// CppMegaNgramHashEmbedding_45122926411843
// MI455X (gfx1250) — matrix operand order repaired and checked by reading, not compiled, not run on hardware
//
#include <hip/hip_runtime.h>

typedef _Float16 half4  __attribute__((ext_vector_type(4)));
typedef _Float16 half8  __attribute__((ext_vector_type(8)));
typedef _Float16 half16 __attribute__((ext_vector_type(16)));
typedef float    float8 __attribute__((ext_vector_type(8)));

#define BATCH   4
#define SEQ     4096
#define NPOS    (BATCH * SEQ)   // 16384
#define NTAB    16
#define EMB     16
#define KDIM    256             // NTAB * EMB
#define NOUT    2048
#define MAXORD  3
#define BPITCH  264             // LDS row pitch in halfs: 256 + 8 pad (bank shift 4)

// ---------------------------------------------------------------------------
// Kernel 1: convert w_out fp32 [2048, 256] -> f16 (B matrix for WMMA)
// ---------------------------------------------------------------------------
__global__ void wcvt_kernel(const float* __restrict__ w, _Float16* __restrict__ wh) {
    int i = (blockIdx.x * blockDim.x + threadIdx.x) * 4;
    float4 v = *(const float4*)(w + i);
    half4 o;
    o.x = (_Float16)v.x; o.y = (_Float16)v.y;
    o.z = (_Float16)v.z; o.w = (_Float16)v.w;
    *(half4*)(wh + i) = o;
}

// ---------------------------------------------------------------------------
// Kernel 2: n-gram hash + embedding gather, fp32 -> f16, X = [NPOS, 256]
// One thread per (position, table). Values fit: mult < 2^31, token < 2^16,
// product < 2^47, XOR-combined hash < 2^48 and non-negative -> uint64 math
// matches the reference int64 semantics, including the zero-padded
// per-batch-row shift and the % prime.
// ---------------------------------------------------------------------------
__global__ void hashgather_kernel(const int* __restrict__ tok,
                                  const float* __restrict__ tw,
                                  const int* __restrict__ mults,
                                  const int* __restrict__ bias,
                                  const int* __restrict__ sizes,
                                  const int* __restrict__ offs,
                                  const int* __restrict__ mask,
                                  _Float16* __restrict__ X) {
    int g   = blockIdx.x * blockDim.x + threadIdx.x;   // [0, NPOS*NTAB)
    int pos = g >> 4;
    int t   = g & 15;
    int s   = pos & (SEQ - 1);

    unsigned long long h = 0ull;
#pragma unroll
    for (int p = 0; p < MAXORD; ++p) {
        unsigned long long tv  = (s >= p) ? (unsigned long long)(unsigned)tok[pos - p] : 0ull;
        unsigned long long mu  = (unsigned long long)(unsigned)mults[t * MAXORD + p];
        unsigned long long msk = (unsigned long long)(unsigned)mask[p * NTAB + t];
        h ^= mu * tv * msk;
    }
    h ^= (unsigned long long)(unsigned)bias[t];
    unsigned long long sz  = (unsigned long long)(unsigned)sizes[t];
    unsigned long long idx = h % sz + (unsigned long long)(unsigned)offs[t];

    const float4* src = (const float4*)(tw + idx * EMB);
    _Float16*     dst = X + (size_t)pos * KDIM + t * EMB;
#pragma unroll
    for (int q = 0; q < 4; ++q) {
        float4 v = src[q];
        half4 o;
        o.x = (_Float16)v.x; o.y = (_Float16)v.y;
        o.z = (_Float16)v.z; o.w = (_Float16)v.w;
        *(half4*)(dst + q * 4) = o;
    }
}

// ---------------------------------------------------------------------------
// Kernel 3: WMMA GEMM  out[M=16384, N=2048] = X[M,256] * Wh[N,256]^T
//
// Block = 8 waves sharing one 64-wide N strip; wave w owns M rows
// [mblk*128 + w*16, +16). The B strip (64 x 256 f16 = 32 KB) is staged in
// LDS once per block (padded pitch 264 halfs -> each 32-lane ds_load_b128
// touches every bank exactly twice = minimal). B reads then ride DScnt while
// A rides LOADcnt with a register double buffer, so the per-step WMMA waits
// are a short LDS hop + a partial loadcnt instead of a full VMEM drain.
// Per wave: 8 k-steps x 4 subtiles of v_wmma_f32_16x16x32_f16.
// ---------------------------------------------------------------------------
__global__ __launch_bounds__(256) void gemm_kernel(const _Float16* __restrict__ X,
                                                   const _Float16* __restrict__ Wh,
                                                   float* __restrict__ out) {
    __shared__ _Float16 Bs[64 * BPITCH];   // 33792 bytes

    const int tid   = threadIdx.x;
    const int ntile = blockIdx.x & 31;     // 32 N strips of 64
    const int mblk  = blockIdx.x >> 5;     // 128 M blocks of 128 rows
    const int n0    = ntile * 64;

    // Cooperative fill: 256 threads x 128 B. Global side fully coalesced.
    {
        const int row = tid >> 2;          // 0..63
        const int seg = tid & 3;           // 4 x 64-half segments per row
        const _Float16* src = Wh + (size_t)(n0 + row) * KDIM + seg * 64;
        _Float16*       dst = Bs + row * BPITCH + seg * 64;
#pragma unroll
        for (int q = 0; q < 8; ++q)
            *(half8*)(dst + q * 8) = *(const half8*)(src + q * 8);
    }
    __syncthreads();

    const int lane = tid & 31;
    const int wave = tid >> 5;
    const int lrow = lane & 15;
    const int lhi  = lane >> 4;
    const int m0   = mblk * 128 + wave * 16;

    const _Float16* arow  = X + (size_t)(m0 + lrow) * KDIM;
    // B fragment: lane n holds K [kb+8*lhi,+8)∪[kb+16+8*lhi,+8) of strip row (16j + n),
    // same K order as A.
    const _Float16* bbase = Bs + lrow * BPITCH + 8 * lhi;

    // A 16x32 fragment: lanes 0-15 hold K [kb,kb+8)∪[kb+16,kb+24),
    // lanes 16-31 hold K [kb+8,kb+16)∪[kb+24,kb+32).
    union AF { half16 v; half8 h[2]; };
    AF a_cur, a_nxt;
    a_cur.h[0] = *(const half8*)(arow + 8 * lhi);
    a_cur.h[1] = *(const half8*)(arow + 16 + 8 * lhi);

    float8 acc[4] = {};

#pragma unroll
    for (int ks = 0; ks < 8; ++ks) {
        if (ks < 7) {   // prefetch next A fragment (only LOADcnt user)
            const int kn = (ks + 1) * 32;
            a_nxt.h[0] = *(const half8*)(arow + kn + 8 * lhi);
            a_nxt.h[1] = *(const half8*)(arow + kn + 16 + 8 * lhi);
        }
        const int kb = ks * 32;
        union BF { half16 v; half8 h[2]; } bf[4];
#pragma unroll
        for (int j = 0; j < 4; ++j) {
            const _Float16* p = bbase + j * 16 * BPITCH + kb;
            bf[j].h[0] = *(const half8*)(p);
            bf[j].h[1] = *(const half8*)(p + 16);
        }
#pragma unroll
        for (int j = 0; j < 4; ++j)
            acc[j] = __builtin_amdgcn_wmma_f32_16x16x32_f16(
                false, a_cur.v, false, bf[j].v, (short)0, acc[j], false, false);
        if (ks < 7)
            a_cur = a_nxt;
    }

    // D layout: lane L, VGPR v -> D[v + 8*(L>>4)][L&15]; stores coalesce over lanes.
    float* obase = out + (size_t)(m0 + 8 * lhi) * NOUT + n0 + lrow;
#pragma unroll
    for (int v = 0; v < 8; ++v) {
        obase[(size_t)v * NOUT +  0] = acc[0][v];
        obase[(size_t)v * NOUT + 16] = acc[1][v];
        obase[(size_t)v * NOUT + 32] = acc[2][v];
        obase[(size_t)v * NOUT + 48] = acc[3][v];
    }
}

// ---------------------------------------------------------------------------
extern "C" void kernel_launch(void* const* d_in, const int* in_sizes, int n_in,
                              void* d_out, int out_size, void* d_ws, size_t ws_size,
                              hipStream_t stream) {
    const int*   tok   = (const int*)d_in[0];
    const float* tw    = (const float*)d_in[1];
    const float* w_out = (const float*)d_in[2];
    const int*   mults = (const int*)d_in[3];
    const int*   bias  = (const int*)d_in[4];
    const int*   sizes = (const int*)d_in[5];
    const int*   offs  = (const int*)d_in[6];
    const int*   mask  = (const int*)d_in[7];
    float*       out   = (float*)d_out;

    // Workspace layout: Wh (1 MB, f16 w_out) then X (8 MB, f16 activations).
    _Float16* Wh = (_Float16*)d_ws;
    _Float16* X  = (_Float16*)((char*)d_ws + (size_t)NOUT * KDIM * sizeof(_Float16));

    wcvt_kernel<<<(NOUT * KDIM / 4) / 256, 256, 0, stream>>>(w_out, Wh);
    hashgather_kernel<<<(NPOS * NTAB) / 256, 256, 0, stream>>>(tok, tw, mults, bias,
                                                               sizes, offs, mask, X);
    gemm_kernel<<<(NPOS / 128) * (NOUT / 64), 256, 0, stream>>>(X, Wh, out);
}
